// GCNModel_22033182229078
// MI455X (gfx1250) — matrix operand order repaired and checked by reading, compiled, not run on hardware
//
#include <hip/hip_runtime.h>
#include <hip/hip_bf16.h>

// ---------------------------------------------------------------------------
// GCN forward for MI455X (gfx1250, wave32, WMMA).
//
// Sizes (fixed by reference):
#define N_NODES 100000
#define N_EDGES 1600000
#define CDIM    128
#define N_LAYERS 8
//
// Precision strategy: node features kept in bf16 for WMMA GEMMs (f32
// accumulate), scatter/aggregation path kept in f32 with hardware
// global_atomic_add_f32 (entire hs/agg working set ~102MB fits in 192MB L2).
// Weights staged in LDS per workgroup (32KB of 320KB/WGP) -> ds_load_b128
// feeds the WMMA B operand instead of redundant global fetches.
// ---------------------------------------------------------------------------

typedef __attribute__((ext_vector_type(16))) __bf16 v16bf;
typedef __attribute__((ext_vector_type(8)))  __bf16 v8bf;
typedef __attribute__((ext_vector_type(8)))  float  v8f;

// ---------------------------------------------------------------------------
// Pack a 128x128 f32 weight (row = K, col = N) into the per-lane bf16 B-matrix
// layout used here for V_WMMA_F32_16X16X32_BF16 (contiguous k, not the ISA table order).
// Packed layout: tile = kb*8 + nt  (kb: K-block of 32, nt: N-block of 16)
//   element j of lane L's v16bf:  K = kb*32 + (L>>4)*16 + j,  N = nt*16 + (L&15)
// ---------------------------------------------------------------------------
__global__ __launch_bounds__(256) void pack_weight_k(const float* __restrict__ src,
                                                     __bf16* __restrict__ dst) {
    int tid = blockIdx.x * blockDim.x + threadIdx.x;   // 0 .. 16383
    if (tid >= 128 * 128) return;
    int j    = tid & 15;
    int lane = (tid >> 4) & 31;
    int tile = tid >> 9;          // 0..31
    int kb   = tile >> 3;
    int nt   = tile & 7;
    int k = kb * 32 + ((lane >> 4) << 4) + j;
    int n = nt * 16 + (lane & 15);
    dst[tid] = (__bf16)src[k * 128 + n];
}

// ---------------------------------------------------------------------------
// Degree / normalization
// ---------------------------------------------------------------------------
__global__ __launch_bounds__(256) void zero_i32_k(int* p, int n) {
    int i = blockIdx.x * blockDim.x + threadIdx.x;
    if (i < n) p[i] = 0;
}
__global__ __launch_bounds__(256) void degree_k(const int* __restrict__ dst, int* deg, int ne) {
    int e = blockIdx.x * blockDim.x + threadIdx.x;
    if (e < ne) atomicAdd(&deg[dst[e]], 1);
}
__global__ __launch_bounds__(256) void dis_k(const int* __restrict__ deg, float* dis, int n) {
    int i = blockIdx.x * blockDim.x + threadIdx.x;
    if (i < n) dis[i] = rsqrtf((float)(deg[i] + 1));
}

// ---------------------------------------------------------------------------
// Preproc: h0 = x@Wp+bp (8->32), ni = relu(h0@W1+b1), h = relu(h0@W2+b2)
// (tiny FLOP count -> plain VALU, one node per thread; weights broadcast via L2)
// ---------------------------------------------------------------------------
__global__ __launch_bounds__(256) void preproc_k(
    const float* __restrict__ x,
    const float* __restrict__ Wp, const float* __restrict__ bp,
    const float* __restrict__ W1, const float* __restrict__ b1,
    const float* __restrict__ W2, const float* __restrict__ b2,
    __bf16* __restrict__ ni, __bf16* __restrict__ h, int n) {
    int i = blockIdx.x * blockDim.x + threadIdx.x;
    if (i >= n) return;
    float xv[8];
    #pragma unroll
    for (int j = 0; j < 8; ++j) xv[j] = x[(size_t)i * 8 + j];
    float h0[32];
    #pragma unroll
    for (int c = 0; c < 32; ++c) {
        float s = bp[c];
        #pragma unroll
        for (int j = 0; j < 8; ++j) s += xv[j] * Wp[j * 32 + c];
        h0[c] = s;
    }
    for (int c = 0; c < 128; ++c) {
        float s1 = b1[c], s2 = b2[c];
        #pragma unroll
        for (int k = 0; k < 32; ++k) {
            float hv = h0[k];
            s1 += hv * W1[k * 128 + c];
            s2 += hv * W2[k * 128 + c];
        }
        ni[(size_t)i * 128 + c] = (__bf16)fmaxf(s1, 0.f);
        h [(size_t)i * 128 + c] = (__bf16)fmaxf(s2, 0.f);
    }
}

// ---------------------------------------------------------------------------
// WMMA GEMM: out[N,128] = A[N,128](bf16) @ B[128,128](packed bf16), f32 accum.
// Packed B is staged into LDS once per 256-thread block (8 waves share it);
// each wave computes a 16-row x 128-col strip: 8 accumulator tiles, 4 K-blocks,
// 32 v_wmma_f32_16x16x32_bf16 per wave, B operands via ds_load_b128.
//   MODE 0: store f32 (out_a)                              [niContrib precompute]
//   MODE 1: val *= dis[row]; store f32 to out_a AND out_b  [hs + agg-init]
//   MODE 2: val += side[row,col] + bias[col]; relu; bf16   [dense fusion / fc1]
// ---------------------------------------------------------------------------
template <int MODE>
__global__ __launch_bounds__(256) void gemm128_k(
    const __bf16* __restrict__ A,
    const __bf16* __restrict__ Bp,
    const float* __restrict__ side,   // MODE1: dis[N]; MODE2: addend [N,128]
    const float* __restrict__ bias,   // MODE2: [128]
    float* __restrict__ out_a,
    float* __restrict__ out_b,
    __bf16* __restrict__ out_bf,
    int nrows) {
    __shared__ __bf16 sB[128 * 128];                 // 32 KB packed weight

    // Cooperative stage: 32KB = 2048 x 16B chunks, 8 chunks per thread.
    {
        const uint4* gsrc = (const uint4*)Bp;
        uint4* ldst = (uint4*)sB;
        #pragma unroll
        for (int t = 0; t < 8; ++t)
            ldst[threadIdx.x + t * 256] = gsrc[threadIdx.x + t * 256];
    }
    __syncthreads();

    const int wave = threadIdx.x >> 5;              // 0..7
    const int lane = threadIdx.x & 31;
    const int row_tile = blockIdx.x * 8 + wave;
    const int m0 = row_tile * 16;
    if (m0 >= nrows) return;

    const int half = lane >> 4;                     // 0/1 : lane group
    const int l15  = lane & 15;
    int m = m0 + l15;
    if (m >= nrows) m = nrows - 1;                  // clamp (stores are guarded)

    v8f acc[8];
    #pragma unroll
    for (int nt = 0; nt < 8; ++nt)
        #pragma unroll
        for (int v = 0; v < 8; ++v) acc[nt][v] = 0.f;

    #pragma unroll
    for (int kb = 0; kb < 4; ++kb) {
        // A layout (contiguous k order, not the ISA table order): lane group 0 holds K[0..15];
        // lane group 1 holds K[16..31] for its row (matches packed B).
        const __bf16* arow = A + (size_t)m * 128 + kb * 32 + half * 16;
        v8bf alo = *(const v8bf*)(arow);
        v8bf ahi = *(const v8bf*)(arow + 8);
        v16bf a;
        #pragma unroll
        for (int i = 0; i < 8; ++i) { a[i] = alo[i]; a[8 + i] = ahi[i]; }

        #pragma unroll
        for (int nt = 0; nt < 8; ++nt) {
            const __bf16* bptr = sB + ((kb * 8 + nt) * 32 + lane) * 16;
            v16bf b = *(const v16bf*)bptr;          // 2x ds_load_b128
            acc[nt] = __builtin_amdgcn_wmma_f32_16x16x32_bf16(
                false, a, false, b, (short)0, acc[nt], false, false);
        }
    }

    // D layout: VGPR v, lane L -> row = m0 + (L>>4)*8 + v, col = nt*16 + (L&15)
    #pragma unroll
    for (int nt = 0; nt < 8; ++nt) {
        const int col = nt * 16 + l15;
        #pragma unroll
        for (int v = 0; v < 8; ++v) {
            const int r = m0 + half * 8 + v;
            if (r < nrows) {
                float val = acc[nt][v];
                if (MODE == 0) {
                    out_a[(size_t)r * 128 + col] = val;
                } else if (MODE == 1) {
                    val *= side[r];
                    out_a[(size_t)r * 128 + col] = val;   // hs  (read by scatter)
                    out_b[(size_t)r * 128 + col] = val;   // agg (self-loop seed)
                } else {
                    val += side[(size_t)r * 128 + col] + bias[col];
                    out_bf[(size_t)r * 128 + col] = (__bf16)fmaxf(val, 0.f);
                }
            }
        }
    }
}

// ---------------------------------------------------------------------------
// Edge scatter: agg[dst] += hs[src].  One thread = one edge x 4 channels.
// f32 atomics resolve in L2 (hs+agg working set ~102MB < 192MB L2).
// ---------------------------------------------------------------------------
__device__ __forceinline__ void atom_add_f32(float* p, float v) {
    __hip_atomic_fetch_add(p, v, __ATOMIC_RELAXED, __HIP_MEMORY_SCOPE_AGENT);
}

__global__ __launch_bounds__(256) void scatter_k(
    const int* __restrict__ src, const int* __restrict__ dst,
    const float* __restrict__ hs, float* __restrict__ agg, int ne) {
    long long tid = (long long)blockIdx.x * blockDim.x + threadIdx.x;
    int e  = (int)(tid >> 5);
    int c4 = ((int)tid & 31) * 4;
    if (e >= ne) return;
    int s = src[e], d = dst[e];
    const float4 v = *(const float4*)(hs + (size_t)s * 128 + c4);
    float* p = agg + (size_t)d * 128 + c4;
    atom_add_f32(p + 0, v.x);
    atom_add_f32(p + 1, v.y);
    atom_add_f32(p + 2, v.z);
    atom_add_f32(p + 3, v.w);
}

// g = relu(agg * dis + bg)  -> bf16 for next WMMA
__global__ __launch_bounds__(256) void post_gcn_k(
    const float* __restrict__ agg, const float* __restrict__ dis,
    const float* __restrict__ bg, __bf16* __restrict__ g, int n) {
    long long idx = (long long)blockIdx.x * blockDim.x + threadIdx.x;
    if (idx >= (long long)n * 128) return;
    int i = (int)(idx >> 7);
    int c = (int)idx & 127;
    float val = agg[idx] * dis[i] + bg[c];
    g[idx] = (__bf16)fmaxf(val, 0.f);
}

// out = hf @ Wf2 + bf2   (128 -> 2)
__global__ __launch_bounds__(256) void final_k(
    const __bf16* __restrict__ hf, const float* __restrict__ Wf2,
    const float* __restrict__ bf2, float* __restrict__ out, int n) {
    int i = blockIdx.x * blockDim.x + threadIdx.x;
    if (i >= n) return;
    float s0 = bf2[0], s1 = bf2[1];
    #pragma unroll 4
    for (int k = 0; k < 128; ++k) {
        float hv = (float)hf[(size_t)i * 128 + k];
        s0 += hv * Wf2[k * 2 + 0];
        s1 += hv * Wf2[k * 2 + 1];
    }
    out[(size_t)i * 2 + 0] = s0;
    out[(size_t)i * 2 + 1] = s1;
}

// ---------------------------------------------------------------------------
extern "C" void kernel_launch(void* const* d_in, const int* in_sizes, int n_in,
                              void* d_out, int out_size, void* d_ws, size_t ws_size,
                              hipStream_t stream) {
    (void)in_sizes; (void)n_in; (void)out_size; (void)ws_size;
    const float* x   = (const float*)d_in[0];
    const int*   ei  = (const int*)  d_in[1];
    const float* Wp  = (const float*)d_in[2];
    const float* bp  = (const float*)d_in[3];
    const float* W1  = (const float*)d_in[4];
    const float* b1  = (const float*)d_in[5];
    const float* W2  = (const float*)d_in[6];
    const float* b2  = (const float*)d_in[7];
    const float* Wg  = (const float*)d_in[8];
    const float* bg  = (const float*)d_in[9];
    const float* Wd  = (const float*)d_in[10];
    const float* bd  = (const float*)d_in[11];
    const float* Wf1 = (const float*)d_in[12];
    const float* bf1 = (const float*)d_in[13];
    const float* Wf2 = (const float*)d_in[14];
    const float* bf2 = (const float*)d_in[15];
    float* out = (float*)d_out;

    const int N = N_NODES, NE = N_EDGES;
    const int* e_src = ei;
    const int* e_dst = ei + NE;

    // --- workspace carving ---
    char* ws = (char*)d_ws;
    auto carve = [&](size_t bytes) -> char* {
        char* p = ws;
        ws += (bytes + 255) & ~(size_t)255;
        return p;
    };
    const size_t PACKED = (size_t)128 * 128 * sizeof(__bf16);   // 32 KB
    __bf16* pWg    = (__bf16*)carve(PACKED);
    __bf16* pWdT   = (__bf16*)carve(PACKED);   // Wd rows   0..127 (ni half)
    __bf16* pWdB   = (__bf16*)carve(PACKED);   // Wd rows 128..255 (g half)
    __bf16* pWf1T  = (__bf16*)carve(PACKED);
    __bf16* pWf1B  = (__bf16*)carve(PACKED);
    int*    deg    = (int*)   carve((size_t)N * 4);
    float*  dis    = (float*) carve((size_t)N * 4);
    __bf16* ni_b   = (__bf16*)carve((size_t)N * 128 * 2);
    __bf16* h_b    = (__bf16*)carve((size_t)N * 128 * 2);
    __bf16* g_b    = (__bf16*)carve((size_t)N * 128 * 2);
    float*  hs     = (float*) carve((size_t)N * 128 * 4);
    float*  agg    = (float*) carve((size_t)N * 128 * 4);
    float*  niD    = (float*) carve((size_t)N * 128 * 4);  // ni @ Wd[:128]
    float*  niF    = (float*) carve((size_t)N * 128 * 4);  // ni @ Wf1[:128]

    const int B256 = 256;
    const int gN      = (N + B256 - 1) / B256;
    const int gPack   = (128 * 128 + B256 - 1) / B256;
    const int gEdge   = (NE + B256 - 1) / B256;
    const int gTiles  = ((N + 15) / 16 + 7) / 8;                       // 8 waves/block
    const int gScat   = (int)(((long long)NE * 32 + B256 - 1) / B256);
    const int gNC     = (int)(((long long)N * 128 + B256 - 1) / B256);

    // 1) Repack weights into WMMA B-layout (bf16).
    pack_weight_k<<<gPack, B256, 0, stream>>>(Wg,              pWg);
    pack_weight_k<<<gPack, B256, 0, stream>>>(Wd,              pWdT);
    pack_weight_k<<<gPack, B256, 0, stream>>>(Wd  + 128 * 128, pWdB);
    pack_weight_k<<<gPack, B256, 0, stream>>>(Wf1,             pWf1T);
    pack_weight_k<<<gPack, B256, 0, stream>>>(Wf1 + 128 * 128, pWf1B);

    // 2) Degrees and deg^-1/2 (self-loop included).
    zero_i32_k<<<gN, B256, 0, stream>>>(deg, N);
    degree_k  <<<gEdge, B256, 0, stream>>>(e_dst, deg, NE);
    dis_k     <<<gN, B256, 0, stream>>>(deg, dis, N);

    // 3) Preproc: ni, h (bf16).
    preproc_k<<<gN, B256, 0, stream>>>(x, Wp, bp, W1, b1, W2, b2, ni_b, h_b, N);

    // 4) Layer-invariant halves of the fused dense layers (WMMA, plain f32).
    gemm128_k<0><<<gTiles, B256, 0, stream>>>(ni_b, pWdT,  nullptr, nullptr, niD, nullptr, nullptr, N);
    gemm128_k<0><<<gTiles, B256, 0, stream>>>(ni_b, pWf1T, nullptr, nullptr, niF, nullptr, nullptr, N);

    // 5) 8 tied GCN layers.
    for (int layer = 0; layer < N_LAYERS; ++layer) {
        // hs = (h @ Wg) * dis ; agg seeded with hs (self-loop term).
        gemm128_k<1><<<gTiles, B256, 0, stream>>>(h_b, pWg, dis, nullptr, hs, agg, nullptr, N);
        // agg[dst] += hs[src]
        scatter_k<<<gScat, B256, 0, stream>>>(e_src, e_dst, hs, agg, NE);
        // g = relu(agg * dis + bg)
        post_gcn_k<<<gNC, B256, 0, stream>>>(agg, dis, bg, g_b, N);
        // h = relu(g @ Wd[128:] + niD + bd)
        gemm128_k<2><<<gTiles, B256, 0, stream>>>(g_b, pWdB, niD, bd, nullptr, nullptr, h_b, N);
    }

    // 6) h = relu(h @ Wf1[128:] + niF + bf1)   (reuse g_b as output)
    gemm128_k<2><<<gTiles, B256, 0, stream>>>(h_b, pWf1B, niF, bf1, nullptr, nullptr, g_b, N);

    // 7) out = h @ Wf2 + bf2
    final_k<<<gN, B256, 0, stream>>>(g_b, Wf2, bf2, out, N);
}
